// Linear_extractor_cluster_16011638079510
// MI455X (gfx1250) — matrix operand order repaired and checked by reading, not compiled, not run on hardware
//
#include <hip/hip_runtime.h>
#include <hip/hip_bf16.h>

typedef __attribute__((ext_vector_type(8)))  __bf16 v8bf;
typedef __attribute__((ext_vector_type(16))) __bf16 v16bf;
typedef __attribute__((ext_vector_type(8)))  float  v8f;

#define GAS __attribute__((address_space(1)))
typedef const v8bf   GAS* g_v8bf_cp;
typedef const __bf16 GAS* g_bf16_cp;
typedef const float  GAS* g_f32_cp;
typedef float        GAS* g_f32_p;
typedef const int    GAS* g_i32_cp;

constexpr int Bc  = 16;
constexpr int Tc  = 512;
constexpr int Cc  = 128;
constexpr int Ec  = 8;
constexpr int DMc = 512;
constexpr int Hc  = 256;

// ---------------------------------------------------------------------------
// Kernel 1: series decomposition (moving average k=25, edge-replicated) and
// transpose to (B, C, T) bf16 layout for WMMA fragment loads.
// One block per (b, c) row.
// ---------------------------------------------------------------------------
__global__ __launch_bounds__(256) void decomp_kernel(
    const float* __restrict__ x, __bf16* __restrict__ Sbf, __bf16* __restrict__ Tbf)
{
  __shared__ float s_x[Tc];
  const int b = blockIdx.x / Cc;
  const int c = blockIdx.x % Cc;
  const int tid = threadIdx.x;

  const float* xb = x + ((size_t)b * Tc) * Cc + c;
  for (int t = tid; t < Tc; t += 256) s_x[t] = xb[(size_t)t * Cc];
  __syncthreads();

  __bf16* Sr = Sbf + ((size_t)b * Cc + c) * Tc;
  __bf16* Tr = Tbf + ((size_t)b * Cc + c) * Tc;
  for (int t = tid; t < Tc; t += 256) {
    float s = 0.f;
#pragma unroll
    for (int j = -12; j <= 12; ++j) {
      int u = t + j;
      u = u < 0 ? 0 : (u > Tc - 1 ? Tc - 1 : u);
      s += s_x[u];
    }
    const float tr = s * (1.0f / 25.0f);
    Sr[t] = (__bf16)(s_x[t] - tr);
    Tr[t] = (__bf16)tr;
  }
}

// ---------------------------------------------------------------------------
// Kernel 2: f32 -> bf16 weight conversion (grid-stride).
// ---------------------------------------------------------------------------
__global__ __launch_bounds__(256) void cvt_kernel(
    const float* __restrict__ a, __bf16* __restrict__ o, int n)
{
  for (int i = blockIdx.x * 256 + threadIdx.x; i < n; i += gridDim.x * 256)
    o[i] = (__bf16)a[i];
}

// ---------------------------------------------------------------------------
// Kernel 3: gating MLP + softmax + top-2 + CV^2 loss. Single workgroup.
// ---------------------------------------------------------------------------
__global__ __launch_bounds__(256) void gate_kernel(
    const float* __restrict__ x, const float* __restrict__ w1,
    const float* __restrict__ w2, const int* __restrict__ loss_coef,
    float* __restrict__ gates, int* __restrict__ gidx, float* __restrict__ loss_out)
{
  __shared__ float s_mean[Bc * Tc];   // 32 KB
  __shared__ float s_hid[Bc * Hc];    // 16 KB
  __shared__ float s_log[Bc * Ec];
  __shared__ float s_gates[Bc * Ec];
  const int tid = threadIdx.x;

  // mean over channel dim
  for (int r = tid; r < Bc * Tc; r += 256) {
    const float* xr = x + (size_t)r * Cc;
    float s = 0.f;
    for (int c = 0; c < Cc; ++c) s += xr[c];
    s_mean[r] = s * (1.0f / Cc);
  }
  __syncthreads();

  // hidden = relu(mean_x @ w1^T)
  for (int i = tid; i < Bc * Hc; i += 256) {
    const int bb = i / Hc, h = i % Hc;
    const float* mr = s_mean + bb * Tc;
    const float* wr = w1 + (size_t)h * Tc;
    float s = 0.f;
    for (int t = 0; t < Tc; ++t) s += mr[t] * wr[t];
    s_hid[i] = s > 0.f ? s : 0.f;
  }
  __syncthreads();

  // logits = hidden @ w2^T
  for (int i = tid; i < Bc * Ec; i += 256) {
    const int bb = i / Ec, e = i % Ec;
    const float* hr = s_hid + bb * Hc;
    const float* wr = w2 + (size_t)e * Hc;
    float s = 0.f;
    for (int h = 0; h < Hc; ++h) s += hr[h] * wr[h];
    s_log[i] = s;
  }
  __syncthreads();

  // softmax + top-2 per batch
  if (tid < Bc) {
    float p[Ec];
    const float* lr = s_log + tid * Ec;
    float m = lr[0];
    for (int e = 1; e < Ec; ++e) m = lr[e] > m ? lr[e] : m;
    float sum = 0.f;
    for (int e = 0; e < Ec; ++e) { p[e] = __expf(lr[e] - m); sum += p[e]; }
    for (int e = 0; e < Ec; ++e) p[e] /= sum;
    int i0 = 0;
    for (int e = 1; e < Ec; ++e) if (p[e] > p[i0]) i0 = e;
    int i1 = (i0 == 0) ? 1 : 0;
    for (int e = 0; e < Ec; ++e) if (e != i0 && p[e] > p[i1]) i1 = e;
    const float denom = p[i0] + p[i1] + 1e-6f;
    const float g0 = p[i0] / denom, g1 = p[i1] / denom;
    gates[tid * 2 + 0] = g0; gates[tid * 2 + 1] = g1;
    gidx[tid * 2 + 0] = i0;  gidx[tid * 2 + 1] = i1;
    for (int e = 0; e < Ec; ++e) s_gates[tid * Ec + e] = 0.f;
    s_gates[tid * Ec + i0] = g0;
    s_gates[tid * Ec + i1] = g1;
  }
  __syncthreads();

  if (tid == 0) {
    float imp[Ec], ld[Ec];
    for (int e = 0; e < Ec; ++e) {
      float si = 0.f, sl = 0.f;
      for (int b = 0; b < Bc; ++b) {
        const float v = s_gates[b * Ec + e];
        si += v;
        sl += (v > 0.f) ? 1.f : 0.f;
      }
      imp[e] = si; ld[e] = sl;
    }
    float loss = 0.f;
    for (int k = 0; k < 2; ++k) {
      const float* v = k == 0 ? imp : ld;
      float mean = 0.f;
      for (int e = 0; e < Ec; ++e) mean += v[e];
      mean *= (1.0f / Ec);
      float var = 0.f;
      for (int e = 0; e < Ec; ++e) { const float d = v[e] - mean; var += d * d; }
      var *= (1.0f / (Ec - 1));
      loss += var / (mean * mean + 1e-10f);
    }
    loss_out[0] = loss * (float)loss_coef[0];
  }
}

// ---------------------------------------------------------------------------
// Kernel 4: top-2 sparse MoE GEMM with v_wmma_f32_16x16x32_bf16.
//   y[b,d,c] = sum_{k<2} g[b,k] * ( sum_t S[b,c,t]*Ws[e_k,d,t]
//                                 + sum_t Tn[b,c,t]*Wt[e_k,d,t]
//                                 + bs[e_k,d] + bt[e_k,d] )
// A-fragments (M=d) come from weights, B-fragments (N=c) from seasonal/trend.
// All hot-path memory traffic goes through address_space(1) pointers so the
// backend emits global_load_b128/global_store_b32 (LOADcnt only) instead of
// flat ops that also tick DScnt.
// Block tile: 128(d) x 64(c), 8 waves each owning 32x32 (4 accumulators).
// ---------------------------------------------------------------------------
__global__ __launch_bounds__(256) void moe_gemm_kernel(
    const __bf16* __restrict__ Sbf_, const __bf16* __restrict__ Tbf_,
    const __bf16* __restrict__ Wsbf_, const __bf16* __restrict__ Wtbf_,
    const float* __restrict__ bs_, const float* __restrict__ bt_,
    const float* __restrict__ gates_, const int* __restrict__ gidx_,
    float* __restrict__ y_)
{
  // hoist everything into the global address space
  g_bf16_cp Sbf  = (g_bf16_cp)Sbf_;
  g_bf16_cp Tbf  = (g_bf16_cp)Tbf_;
  g_bf16_cp Wsbf = (g_bf16_cp)Wsbf_;
  g_bf16_cp Wtbf = (g_bf16_cp)Wtbf_;
  g_f32_cp  bs   = (g_f32_cp)bs_;
  g_f32_cp  bt   = (g_f32_cp)bt_;
  g_f32_cp  gates = (g_f32_cp)gates_;
  g_i32_cp  gidx  = (g_i32_cp)gidx_;
  g_f32_p   y     = (g_f32_p)y_;

  const int b      = blockIdx.z;
  const int dblock = blockIdx.x * 128;
  const int cblock = blockIdx.y * 64;
  const int lane   = threadIdx.x & 31;
  const int w      = threadIdx.x >> 5;
  const int wd     = (w & 3) * 32;   // wave's d offset within block tile
  const int wc     = (w >> 2) * 32;  // wave's c offset within block tile
  const int lm     = lane & 15;
  const int lhi    = lane >> 4;

  v8f tot[2][2];
#pragma unroll
  for (int i = 0; i < 2; ++i)
#pragma unroll
    for (int j = 0; j < 2; ++j) tot[i][j] = (v8f)(0.f);

#pragma unroll 1
  for (int ke = 0; ke < 2; ++ke) {
    const int   e  = gidx[b * 2 + ke];
    const float ge = gates[b * 2 + ke];

    v8f acc[2][2];
#pragma unroll
    for (int i = 0; i < 2; ++i)
#pragma unroll
      for (int j = 0; j < 2; ++j) acc[i][j] = (v8f)(0.f);

    g_bf16_cp Wbase[2] = { Wsbf + (size_t)e * DMc * Tc,
                           Wtbf + (size_t)e * DMc * Tc };
    g_bf16_cp Xbase[2] = { Sbf + (size_t)b * Cc * Tc,
                           Tbf + (size_t)b * Cc * Tc };

#pragma unroll 1
    for (int s = 0; s < 2; ++s) {
      g_bf16_cp Wp = Wbase[s];
      g_bf16_cp Xp = Xbase[s];
#pragma unroll 4
      for (int t = 0; t < Tc; t += 32) {
        v16bf afrag[2], bfrag[2];
#pragma unroll
        for (int i = 0; i < 2; ++i) {
          // 16-bit A 16x32 layout: lane holds row m=lane%16; k-halves
          // [lhi*8 .. +7] and [lhi*8+16 .. +23]  -> two 16B global loads
          g_bf16_cp p = Wp + (size_t)(dblock + wd + i * 16 + lm) * Tc + t + lhi * 8;
          v8bf lo = *(g_v8bf_cp)p;
          v8bf hi = *(g_v8bf_cp)(p + 16);
          afrag[i] = __builtin_shufflevector(lo, hi, 0, 1, 2, 3, 4, 5, 6, 7,
                                             8, 9, 10, 11, 12, 13, 14, 15);
        }
#pragma unroll
        for (int j = 0; j < 2; ++j) {
          // 16-bit B 32x16 layout: lane holds col n=lane%16; k-halves
          // [lhi*8 .. +7] and [lhi*8+16 .. +23] -> two 16B global loads
          g_bf16_cp p = Xp + (size_t)(cblock + wc + j * 16 + lm) * Tc + t + lhi * 8;
          v8bf lo = *(g_v8bf_cp)p;
          v8bf hi = *(g_v8bf_cp)(p + 16);
          bfrag[j] = __builtin_shufflevector(lo, hi, 0, 1, 2, 3, 4, 5, 6, 7,
                                             8, 9, 10, 11, 12, 13, 14, 15);
        }
#pragma unroll
        for (int i = 0; i < 2; ++i)
#pragma unroll
          for (int j = 0; j < 2; ++j)
            acc[i][j] = __builtin_amdgcn_wmma_f32_16x16x32_bf16(
                false, afrag[i], false, bfrag[j], (short)0, acc[i][j], false, false);
      }
    }

#pragma unroll
    for (int i = 0; i < 2; ++i)
#pragma unroll
      for (int j = 0; j < 2; ++j)
#pragma unroll
        for (int r = 0; r < 8; ++r) tot[i][j][r] += ge * acc[i][j][r];
  }

  // fused bias + store: lane pattern -> d = base + r + 8*lhi, c = base + lane%16
  const float g0 = gates[b * 2 + 0], g1 = gates[b * 2 + 1];
  const int   e0 = gidx[b * 2 + 0],  e1 = gidx[b * 2 + 1];
#pragma unroll
  for (int i = 0; i < 2; ++i) {
#pragma unroll
    for (int r = 0; r < 8; ++r) {
      const int d = dblock + wd + i * 16 + r + lhi * 8;
      const float bias = g0 * (bs[e0 * DMc + d] + bt[e0 * DMc + d]) +
                         g1 * (bs[e1 * DMc + d] + bt[e1 * DMc + d]);
#pragma unroll
      for (int j = 0; j < 2; ++j) {
        const int c = cblock + wc + j * 16 + lm;
        y[((size_t)b * DMc + d) * Cc + c] = tot[i][j][r] + bias;
      }
    }
  }
}

// ---------------------------------------------------------------------------
extern "C" void kernel_launch(void* const* d_in, const int* in_sizes, int n_in,
                              void* d_out, int out_size, void* d_ws, size_t ws_size,
                              hipStream_t stream) {
  const float* x       = (const float*)d_in[0];
  const float* gate_w1 = (const float*)d_in[1];
  const float* gate_w2 = (const float*)d_in[2];
  const float* Ws      = (const float*)d_in[3];
  const float* bs      = (const float*)d_in[4];
  const float* Wt      = (const float*)d_in[5];
  const float* bt      = (const float*)d_in[6];
  const int*   lcoef   = (const int*)d_in[7];

  float* y = (float*)d_out;
  float* loss_out = y + (size_t)Bc * DMc * Cc;

  // workspace layout
  __bf16* Sbf  = (__bf16*)d_ws;                       // B*C*T
  __bf16* Tbf  = Sbf  + (size_t)Bc * Cc * Tc;         // B*C*T
  __bf16* Wsbf = Tbf  + (size_t)Bc * Cc * Tc;         // E*DM*T
  __bf16* Wtbf = Wsbf + (size_t)Ec * DMc * Tc;        // E*DM*T
  float*  gts  = (float*)(Wtbf + (size_t)Ec * DMc * Tc);
  int*    gix  = (int*)(gts + Bc * 2);

  // 1) decomposition + transpose + bf16
  decomp_kernel<<<Bc * Cc, 256, 0, stream>>>(x, Sbf, Tbf);

  // 2) weight conversion
  const int nw = Ec * DMc * Tc;
  cvt_kernel<<<1024, 256, 0, stream>>>(Ws, Wsbf, nw);
  cvt_kernel<<<1024, 256, 0, stream>>>(Wt, Wtbf, nw);

  // 3) gating + loss
  gate_kernel<<<1, 256, 0, stream>>>(x, gate_w1, gate_w2, lcoef, gts, gix, loss_out);

  // 4) top-2 sparse expert GEMM + combine
  dim3 grid(DMc / 128, Cc / 64, Bc);
  moe_gemm_kernel<<<grid, 256, 0, stream>>>(Sbf, Tbf, Wsbf, Wtbf, bs, bt, gts, gix, y);
}
